// ReflectiveMultiHeadAttention_9354438771233
// MI455X (gfx1250) — matrix operand order repaired and checked by reading, not compiled, not run on hardware
//
#include <hip/hip_runtime.h>
#include <hip/hip_bf16.h>

#define D_MODEL 1024
#define NH 16
#define DK 64
#define BB 2
#define SS 2048
#define MROWS (BB * SS)          // 4096

typedef __attribute__((ext_vector_type(16))) _Float16 v16h;
typedef __attribute__((ext_vector_type(8)))  _Float16 v8h;
typedef __attribute__((ext_vector_type(8)))  float    v8f;

// Load a v16h WMMA operand from two contiguous 16B LDS chunks.
__device__ inline v16h ld16(const _Float16* p0, const _Float16* p1) {
    union { v16h v; v8h h[2]; } u;
    u.h[0] = *(const v8h*)p0;
    u.h[1] = *(const v8h*)p1;
    return u.v;
}

__device__ inline v8f wmma_f16(v16h a, v16h b, v8f c) {
    return __builtin_amdgcn_wmma_f32_16x16x32_f16(false, a, false, b, (short)0, c, false, false);
}

// Async 16B copy global -> LDS (ASYNCcnt-tracked, VFLAT encoding, gfx1250).
// VDST = 32-bit LDS byte address (low 32 bits of the flat shared pointer),
// VADDR = 64-bit global address.
__device__ inline void async_cp16(const _Float16* gsrc, _Float16* ldst) {
    unsigned lds_off = (unsigned)(size_t)ldst;
    asm volatile("global_load_async_to_lds_b128 %0, %1, off"
                 :: "v"(lds_off), "v"(gsrc)
                 : "memory");
}

__device__ inline void wait_async() {
    asm volatile("s_wait_asynccnt 0x0" ::: "memory");
}

// ---------------------------------------------------------------------------
// gate[b,h,s] = sigmoid(logistics[b,s,:] . gate_W[h,:] + gate_b[h])
// ---------------------------------------------------------------------------
__global__ __launch_bounds__(256) void gate_kernel(const float* __restrict__ logi,
                                                   const float* __restrict__ gW,
                                                   const float* __restrict__ gb,
                                                   float* __restrict__ gate) {
    int id = blockIdx.x * 256 + threadIdx.x;          // BB*NH*SS = 65536
    int s  = id & (SS - 1);
    int h  = (id >> 11) & (NH - 1);
    int bb = id >> 15;
    const float* lp = logi + ((size_t)bb * SS + s) * DK;
    const float* wp = gW + h * DK;
    float acc = gb[h];
#pragma unroll 8
    for (int j = 0; j < DK; j++) acc += lp[j] * wp[j];
    gate[id] = 1.0f / (1.0f + __expf(-acc));
}

// ---------------------------------------------------------------------------
// OUT_f16[b,h,s,d] = (X[m,:] @ W[n,:]^T + bias[n]) * gate * scale  (f16 store)
// ---------------------------------------------------------------------------
__global__ __launch_bounds__(256) void proj_gemm(const float* __restrict__ X,
                                                 const float* __restrict__ W,
                                                 const float* __restrict__ bias,
                                                 const float* __restrict__ gate,
                                                 float scale,
                                                 _Float16* __restrict__ out) {
    __shared__ __align__(16) _Float16 a_lds[128 * 64];
    __shared__ __align__(16) _Float16 b_lds[128 * 64];

    const int m0 = blockIdx.x * 128;
    const int n0 = blockIdx.y * 128;
    const int t = threadIdx.x;
    const int lane = t & 31, wid = t >> 5;
    const int hh = lane >> 4, l15 = lane & 15;

    v8f zero = {};
    v8f acc[8];
#pragma unroll
    for (int i = 0; i < 8; i++) acc[i] = zero;

    for (int kt = 0; kt < D_MODEL; kt += 64) {
        {
            const int row = t >> 1, seg = (t & 1) * 32;
            const float* srcA = X + (size_t)(m0 + row) * D_MODEL + kt + seg;
            _Float16* dstA = a_lds + row * 64 + seg;
            const float* srcB = W + (size_t)(n0 + row) * D_MODEL + kt + seg;
            _Float16* dstB = b_lds + row * 64 + seg;
#pragma unroll
            for (int j = 0; j < 32; j += 4) {
                float4 va = *(const float4*)(srcA + j);
                dstA[j + 0] = (_Float16)va.x; dstA[j + 1] = (_Float16)va.y;
                dstA[j + 2] = (_Float16)va.z; dstA[j + 3] = (_Float16)va.w;
                float4 vb = *(const float4*)(srcB + j);
                dstB[j + 0] = (_Float16)vb.x; dstB[j + 1] = (_Float16)vb.y;
                dstB[j + 2] = (_Float16)vb.z; dstB[j + 3] = (_Float16)vb.w;
            }
        }
        __syncthreads();
#pragma unroll
        for (int kk = 0; kk < 64; kk += 32) {
            const _Float16* ap = a_lds + (wid * 16 + l15) * 64 + kk + hh * 16;
            v16h A = ld16(ap, ap + 8);
            const _Float16* bp0 = b_lds + l15 * 64 + kk + hh * 16;
            v16h Bc = ld16(bp0, bp0 + 8);
#pragma unroll
            for (int ct = 0; ct < 8; ct++) {
                v16h Bn = Bc;
                if (ct < 7) {   // prefetch next B tile before current WMMA
                    const _Float16* bp = b_lds + ((ct + 1) * 16 + l15) * 64 + kk + hh * 16;
                    Bn = ld16(bp, bp + 8);
                }
                acc[ct] = wmma_f16(A, Bc, acc[ct]);
                Bc = Bn;
            }
        }
        __syncthreads();
    }

#pragma unroll
    for (int ct = 0; ct < 8; ct++) {
        const int n = n0 + ct * 16 + l15;
        const int h = n >> 6, d = n & 63;
        const float bn = bias[n];
#pragma unroll
        for (int r = 0; r < 8; r++) {
            const int m = m0 + wid * 16 + r + hh * 8;
            const int bb = m >> 11, s = m & (SS - 1);
            float g = 1.0f;
            if (gate) g = gate[((size_t)(bb * NH + h) * SS) + s];
            float val = (acc[ct][r] + bn) * g * scale;
            out[((size_t)(bb * NH + h) * SS + s) * DK + d] = (_Float16)val;
        }
    }
}

// ---------------------------------------------------------------------------
// VT[b,h,d,s] = V[b,h,s,d]  (one-shot f16 transpose so attention can stage
// V tiles with verbatim async copies)
// ---------------------------------------------------------------------------
__global__ __launch_bounds__(256) void vtrans_kernel(const _Float16* __restrict__ V,
                                                     _Float16* __restrict__ VT) {
    __shared__ _Float16 tl[64 * 65];
    const int s0 = blockIdx.x * 64;
    const int bh = blockIdx.y;
    const _Float16* Vh = V + (size_t)bh * SS * DK;
    _Float16* Vth = VT + (size_t)bh * DK * SS;
    const int t = threadIdx.x;

    const int r = t & 63, cseg = (t >> 6) * 16;
    union { uint4 u; _Float16 h[8]; } a0, a1;
    a0.u = *(const uint4*)(Vh + (size_t)(s0 + r) * DK + cseg);
    a1.u = *(const uint4*)(Vh + (size_t)(s0 + r) * DK + cseg + 8);
    _Float16* dl = tl + r * 65 + cseg;
#pragma unroll
    for (int j = 0; j < 8; j++) { dl[j] = a0.h[j]; dl[8 + j] = a1.h[j]; }
    __syncthreads();

    const int d = t & 63, sseg = (t >> 6) * 16;
    v8h o0, o1;
#pragma unroll
    for (int j = 0; j < 8; j++) {
        o0[j] = tl[(sseg + j) * 65 + d];
        o1[j] = tl[(sseg + 8 + j) * 65 + d];
    }
    _Float16* dst = Vth + (size_t)d * SS + s0 + sseg;
    *(v8h*)dst = o0;
    *(v8h*)(dst + 8) = o1;
}

// ---------------------------------------------------------------------------
// Two-pass flash attention with materialized softmax weights.
// grid (S/128, B*H), 256 threads (8 waves, each owns 16 query rows)
// K and VT tiles are double-buffered via async global->LDS copies.
// ---------------------------------------------------------------------------
__global__ __launch_bounds__(256) void attn_kernel(const _Float16* __restrict__ Qf,
                                                   const _Float16* __restrict__ Kf,
                                                   const _Float16* __restrict__ VT,
                                                   float* __restrict__ weights,
                                                   _Float16* __restrict__ attn_out) {
    __shared__ __align__(16) _Float16 q_lds[128 * 64];
    __shared__ __align__(16) _Float16 k_lds[2][128 * 64];
    __shared__ __align__(16) _Float16 vt_lds[2][64 * 128];
    __shared__ __align__(16) _Float16 w_lds[8][16 * 128];

    const int q0 = blockIdx.x * 128;
    const int bh = blockIdx.y;
    const _Float16* Qh = Qf + (size_t)bh * SS * DK;
    const _Float16* Kh = Kf + (size_t)bh * SS * DK;
    const _Float16* VTh = VT + (size_t)bh * DK * SS;
    float* Wh = weights + (size_t)bh * SS * SS;

    const int t = threadIdx.x;
    const int lane = t & 31, wid = t >> 5;
    const int hh = lane >> 4, l15 = lane & 15;

    // --- staging helpers (each thread moves 4x16B) ---
    auto stage_k = [&](int it, _Float16* dst) {
        const _Float16* src = Kh + (size_t)(it * 128) * DK;   // tile is contiguous
#pragma unroll
        for (int j = 0; j < 4; j++) async_cp16(src + t * 32 + j * 8, dst + t * 32 + j * 8);
    };
    auto stage_vt = [&](int it, _Float16* dst) {
        const int k0 = it * 128;
        const int row = t >> 2;                  // 0..63 (d)
#pragma unroll
        for (int j = 0; j < 4; j++) {
            const int cc = ((t & 3) * 4 + j) * 8;  // 0..120
            async_cp16(VTh + (size_t)row * SS + k0 + cc, dst + row * 128 + cc);
        }
    };

    // stage Q block (pre-scaled by 1/sqrt(dk)) + first K tile
    {
        const _Float16* src = Qh + (size_t)q0 * DK;
#pragma unroll
        for (int j = 0; j < 4; j++) async_cp16(src + t * 32 + j * 8, q_lds + t * 32 + j * 8);
    }
    stage_k(0, k_lds[0]);
    wait_async();
    __syncthreads();

    v8f zero = {};
    float m_r[8], l_r[8];
#pragma unroll
    for (int r = 0; r < 8; r++) { m_r[r] = -3.0e38f; l_r[r] = 0.0f; }

    // ---- pass A: online max / sum over all key tiles ----
    for (int it = 0; it < SS / 128; it++) {
        const _Float16* cur = k_lds[it & 1];
        if (it + 1 < SS / 128) stage_k(it + 1, k_lds[(it + 1) & 1]);

        v8f st[8];
#pragma unroll
        for (int i = 0; i < 8; i++) st[i] = zero;
#pragma unroll
        for (int kk = 0; kk < 64; kk += 32) {
            const _Float16* ap = q_lds + (wid * 16 + l15) * 64 + kk + hh * 16;
            v16h A = ld16(ap, ap + 8);
            const _Float16* bp0 = cur + l15 * 64 + kk + hh * 16;
            v16h Bc = ld16(bp0, bp0 + 8);
#pragma unroll
            for (int ct = 0; ct < 8; ct++) {
                v16h Bn = Bc;
                if (ct < 7) {
                    const _Float16* bp = cur + ((ct + 1) * 16 + l15) * 64 + kk + hh * 16;
                    Bn = ld16(bp, bp + 8);
                }
                st[ct] = wmma_f16(A, Bc, st[ct]);
                Bc = Bn;
            }
        }
#pragma unroll
        for (int r = 0; r < 8; r++) {
            float mx = st[0][r];
#pragma unroll
            for (int ct = 1; ct < 8; ct++) mx = fmaxf(mx, st[ct][r]);
            mx = fmaxf(mx, __shfl_xor(mx, 1, 16));
            mx = fmaxf(mx, __shfl_xor(mx, 2, 16));
            mx = fmaxf(mx, __shfl_xor(mx, 4, 16));
            mx = fmaxf(mx, __shfl_xor(mx, 8, 16));
            const float mnew = fmaxf(m_r[r], mx);
            float ssum = 0.0f;
#pragma unroll
            for (int ct = 0; ct < 8; ct++) ssum += __expf(st[ct][r] - mnew);
            ssum += __shfl_xor(ssum, 1, 16);
            ssum += __shfl_xor(ssum, 2, 16);
            ssum += __shfl_xor(ssum, 4, 16);
            ssum += __shfl_xor(ssum, 8, 16);
            l_r[r] = l_r[r] * __expf(m_r[r] - mnew) + ssum;
            m_r[r] = mnew;
        }
        wait_async();
        __syncthreads();
    }

    float inv_l[8];
#pragma unroll
    for (int r = 0; r < 8; r++) inv_l[r] = 1.0f / l_r[r];

    // ---- pass B: recompute scores, emit weights, accumulate W @ V ----
    v8f oacc[4];
#pragma unroll
    for (int i = 0; i < 4; i++) oacc[i] = zero;

    stage_k(0, k_lds[0]);
    stage_vt(0, vt_lds[0]);
    wait_async();
    __syncthreads();

    for (int it = 0; it < SS / 128; it++) {
        const int k0 = it * 128;
        const _Float16* curk = k_lds[it & 1];
        const _Float16* curv = vt_lds[it & 1];
        if (it + 1 < SS / 128) {
            stage_k(it + 1, k_lds[(it + 1) & 1]);
            stage_vt(it + 1, vt_lds[(it + 1) & 1]);
        }

        v8f st[8];
#pragma unroll
        for (int i = 0; i < 8; i++) st[i] = zero;
#pragma unroll
        for (int kk = 0; kk < 64; kk += 32) {
            const _Float16* ap = q_lds + (wid * 16 + l15) * 64 + kk + hh * 16;
            v16h A = ld16(ap, ap + 8);
            const _Float16* bp0 = curk + l15 * 64 + kk + hh * 16;
            v16h Bc = ld16(bp0, bp0 + 8);
#pragma unroll
            for (int ct = 0; ct < 8; ct++) {
                v16h Bn = Bc;
                if (ct < 7) {
                    const _Float16* bp = curk + ((ct + 1) * 16 + l15) * 64 + kk + hh * 16;
                    Bn = ld16(bp, bp + 8);
                }
                st[ct] = wmma_f16(A, Bc, st[ct]);
                Bc = Bn;
            }
        }

        // normalized weights -> global (single write) and per-wave LDS buffer
        _Float16* wbuf = &w_lds[wid][0];
#pragma unroll
        for (int ct = 0; ct < 8; ct++) {
            const int kcol = ct * 16 + l15;
#pragma unroll
            for (int r = 0; r < 8; r++) {
                const float w = __expf(st[ct][r] - m_r[r]) * inv_l[r];
                const int qv = wid * 16 + r + hh * 8;
                Wh[(size_t)(q0 + qv) * SS + (k0 + kcol)] = w;
                wbuf[(r + hh * 8) * 128 + kcol] = (_Float16)w;
            }
        }

        // attn += W_tile[16x128] @ V_tile[128x64]  (per-wave LDS RAW is in-order)
#pragma unroll
        for (int kk = 0; kk < 128; kk += 32) {
            const _Float16* ap = wbuf + l15 * 128 + kk + hh * 16;
            v16h A = ld16(ap, ap + 8);
            const _Float16* bp0 = curv + l15 * 128 + kk + hh * 16;
            v16h Bc = ld16(bp0, bp0 + 8);
#pragma unroll
            for (int td = 0; td < 4; td++) {
                v16h Bn = Bc;
                if (td < 3) {
                    const _Float16* bp = curv + ((td + 1) * 16 + l15) * 128 + kk + hh * 16;
                    Bn = ld16(bp, bp + 8);
                }
                oacc[td] = wmma_f16(A, Bc, oacc[td]);
                Bc = Bn;
            }
        }
        wait_async();
        __syncthreads();
    }

    // write attn as f16 in [B,S,H*dk] layout for the output projection
    const int bb = bh >> 4, h = bh & (NH - 1);
#pragma unroll
    for (int td = 0; td < 4; td++) {
        const int d = td * 16 + l15;
#pragma unroll
        for (int r = 0; r < 8; r++) {
            const int q = q0 + wid * 16 + r + hh * 8;
            attn_out[((size_t)(bb * SS + q) * NH + h) * DK + d] = (_Float16)oacc[td][r];
        }
    }
}

// ---------------------------------------------------------------------------
// y0 = attn_f16 @ wo^T + bo + residual  (f32 store into d_out)
// ---------------------------------------------------------------------------
__global__ __launch_bounds__(256) void oproj_gemm(const _Float16* __restrict__ Af,
                                                  const float* __restrict__ W,
                                                  const float* __restrict__ bias,
                                                  const float* __restrict__ resid,
                                                  float* __restrict__ out) {
    __shared__ __align__(16) _Float16 a_lds[128 * 64];
    __shared__ __align__(16) _Float16 b_lds[128 * 64];

    const int m0 = blockIdx.x * 128;
    const int n0 = blockIdx.y * 128;
    const int t = threadIdx.x;
    const int lane = t & 31, wid = t >> 5;
    const int hh = lane >> 4, l15 = lane & 15;

    v8f zero = {};
    v8f acc[8];
#pragma unroll
    for (int i = 0; i < 8; i++) acc[i] = zero;

    for (int kt = 0; kt < D_MODEL; kt += 64) {
        {
            const int row = t >> 1, seg = (t & 1) * 32;
            const _Float16* srcA = Af + (size_t)(m0 + row) * D_MODEL + kt + seg;
            _Float16* dstA = a_lds + row * 64 + seg;
#pragma unroll
            for (int j = 0; j < 4; j++) async_cp16(srcA + j * 8, dstA + j * 8);
            const float* srcB = W + (size_t)(n0 + row) * D_MODEL + kt + seg;
            _Float16* dstB = b_lds + row * 64 + seg;
#pragma unroll
            for (int j = 0; j < 32; j += 4) {
                float4 vb = *(const float4*)(srcB + j);
                dstB[j + 0] = (_Float16)vb.x; dstB[j + 1] = (_Float16)vb.y;
                dstB[j + 2] = (_Float16)vb.z; dstB[j + 3] = (_Float16)vb.w;
            }
        }
        wait_async();
        __syncthreads();
#pragma unroll
        for (int kk = 0; kk < 64; kk += 32) {
            const _Float16* ap = a_lds + (wid * 16 + l15) * 64 + kk + hh * 16;
            v16h A = ld16(ap, ap + 8);
            const _Float16* bp0 = b_lds + l15 * 64 + kk + hh * 16;
            v16h Bc = ld16(bp0, bp0 + 8);
#pragma unroll
            for (int ct = 0; ct < 8; ct++) {
                v16h Bn = Bc;
                if (ct < 7) {
                    const _Float16* bp = b_lds + ((ct + 1) * 16 + l15) * 64 + kk + hh * 16;
                    Bn = ld16(bp, bp + 8);
                }
                acc[ct] = wmma_f16(A, Bc, acc[ct]);
                Bc = Bn;
            }
        }
        __syncthreads();
    }

#pragma unroll
    for (int ct = 0; ct < 8; ct++) {
        const int n = n0 + ct * 16 + l15;
        const float bn = bias[n];
#pragma unroll
        for (int r = 0; r < 8; r++) {
            const int m = m0 + wid * 16 + r + hh * 8;
            const size_t idx = (size_t)m * D_MODEL + n;
            out[idx] = acc[ct][r] + bn + resid[idx];
        }
    }
}

// ---------------------------------------------------------------------------
// In-place LayerNorm over rows of y (d_out), D = 1024, one block per row
// ---------------------------------------------------------------------------
__global__ __launch_bounds__(256) void ln_kernel(float* __restrict__ y,
                                                 const float* __restrict__ g,
                                                 const float* __restrict__ b) {
    __shared__ float red[256];
    const int t = threadIdx.x;
    float* yp = y + (size_t)blockIdx.x * D_MODEL;

    float local[4];
    float s = 0.0f;
#pragma unroll
    for (int i = 0; i < 4; i++) { local[i] = yp[t + i * 256]; s += local[i]; }
    red[t] = s; __syncthreads();
    for (int off = 128; off > 0; off >>= 1) {
        if (t < off) red[t] += red[t + off];
        __syncthreads();
    }
    const float mu = red[0] * (1.0f / D_MODEL);
    __syncthreads();

    float vs = 0.0f;
#pragma unroll
    for (int i = 0; i < 4; i++) { float d = local[i] - mu; vs += d * d; }
    red[t] = vs; __syncthreads();
    for (int off = 128; off > 0; off >>= 1) {
        if (t < off) red[t] += red[t + off];
        __syncthreads();
    }
    const float inv = rsqrtf(red[0] * (1.0f / D_MODEL) + 1e-5f);
    __syncthreads();

#pragma unroll
    for (int i = 0; i < 4; i++) {
        const int c = t + i * 256;
        yp[c] = (local[i] - mu) * inv * g[c] + b[c];
    }
}

// ---------------------------------------------------------------------------
extern "C" void kernel_launch(void* const* d_in, const int* in_sizes, int n_in,
                              void* d_out, int out_size, void* d_ws, size_t ws_size,
                              hipStream_t stream) {
    const float* query = (const float*)d_in[0];
    const float* key_  = (const float*)d_in[1];
    const float* value = (const float*)d_in[2];
    const float* logi  = (const float*)d_in[3];
    const float* wq = (const float*)d_in[4];
    const float* bq = (const float*)d_in[5];
    const float* wk = (const float*)d_in[6];
    const float* bk = (const float*)d_in[7];
    const float* wvp = (const float*)d_in[8];
    const float* bv = (const float*)d_in[9];
    const float* wo = (const float*)d_in[10];
    const float* bo = (const float*)d_in[11];
    const float* gW = (const float*)d_in[12];
    const float* gb = (const float*)d_in[13];
    const float* ln_g = (const float*)d_in[14];
    const float* ln_b = (const float*)d_in[15];

    char* ws = (char*)d_ws;
    const size_t tens = (size_t)BB * NH * SS * DK * sizeof(_Float16);  // 8 MB
    _Float16* Qf = (_Float16*)(ws);
    _Float16* Kf = (_Float16*)(ws + tens);
    _Float16* Vf = (_Float16*)(ws + 2 * tens);
    _Float16* VT = (_Float16*)(ws + 3 * tens);
    _Float16* Af = (_Float16*)(ws + 4 * tens);
    float* gate = (float*)(ws + 5 * tens);

    float* y_out = (float*)d_out;
    float* w_out = y_out + (size_t)BB * SS * D_MODEL;

    gate_kernel<<<(BB * NH * SS) / 256, 256, 0, stream>>>(logi, gW, gb, gate);

    dim3 gemm_grid(MROWS / 128, D_MODEL / 128);
    proj_gemm<<<gemm_grid, 256, 0, stream>>>(query, wq, bq, nullptr, 0.125f, Qf); // 1/sqrt(64)
    proj_gemm<<<gemm_grid, 256, 0, stream>>>(key_,  wk, bk, gate,    1.0f,   Kf);
    proj_gemm<<<gemm_grid, 256, 0, stream>>>(value, wvp, bv, nullptr, 1.0f,  Vf);

    vtrans_kernel<<<dim3(SS / 64, BB * NH), 256, 0, stream>>>(Vf, VT);

    attn_kernel<<<dim3(SS / 128, BB * NH), 256, 0, stream>>>(Qf, Kf, VT, w_out, Af);

    oproj_gemm<<<gemm_grid, 256, 0, stream>>>(Af, wo, bo, query, y_out);

    ln_kernel<<<MROWS, 256, 0, stream>>>(y_out, ln_g, ln_b);
}
